// StructuredPrimitiveRAG_77884936945934
// MI455X (gfx1250) — matrix operand order repaired and checked by reading, not compiled, not run on hardware
//
#include <hip/hip_runtime.h>

// ---------------- problem constants ----------------
#define B_   128
#define N_   (1 << 20)
#define E_   128
#define CD_  64
#define TD_  16
#define H_   256
#define K_   16
#define OD_  1
#define PD_  (CD_ + TD_)

// ---------------- tiling ----------------
#define NBLK   256                 // blocks in the similarity pass
#define CHUNK  (N_ / NBLK)         // 4096 corpus rows per block
#define TROWS  32                  // corpus rows per tile
#define TILES  (CHUNK / TROWS)     // 128 tiles
#define DEPTH  3                   // async-to-LDS pipeline depth
#define CPP    32                  // candidates per (block, patient)
#define POOL   (NBLK * CPP)        // 8192 candidates per patient at merge

// ---------------- workspace layout (float elements) ----------------
#define WS_PE   0                          // [128*128] normalized patient emb
#define WS_CSC  (WS_PE + B_ * E_)          // [NBLK*B_*CPP] candidate scores
#define WS_CID  (WS_CSC + NBLK * B_ * CPP) // [NBLK*B_*CPP] candidate indices (int)
// total ~8.5 MB of d_ws

// ---------------- d_out layout (elements, return order) ----------------
#define OUT_OUTCOME 0
#define OUT_SCORES  (B_ * OD_)              // 128
#define OUT_IDX     (OUT_SCORES + B_ * K_)  // 2176
#define OUT_ATTR    (OUT_IDX + B_ * K_)     // 4224
#define OUT_RETR    (OUT_ATTR + B_ * K_)    // 6272 (+128*16*128 = 268416 total)

#define NEG_INF (-__builtin_inff())

typedef __attribute__((ext_vector_type(16))) __bf16          v16bf;
typedef __attribute__((ext_vector_type(16))) unsigned short  v16u;
typedef __attribute__((ext_vector_type(8)))  unsigned short  v8u;
typedef __attribute__((ext_vector_type(8)))  float           v8f;
typedef __attribute__((ext_vector_type(4)))  int             v4i;

typedef __attribute__((address_space(1))) v4i* gv4i_p;
typedef __attribute__((address_space(3))) v4i* lv4i_p;

// ---------------- gfx1250 async global->LDS (guarded) ----------------
#if __has_builtin(__builtin_amdgcn_global_load_async_to_lds_b128)
#define HAVE_ASYNC 1
#else
#define HAVE_ASYNC 0
#endif

__device__ __forceinline__ void gl2lds_b128(const float* g, float* l) {
#if HAVE_ASYNC
  __builtin_amdgcn_global_load_async_to_lds_b128((gv4i_p)g, (lv4i_p)l, 0, 0);
#else
  *(float4*)l = *(const float4*)g;          // synchronous fallback
#endif
}

template <int NWAIT>
__device__ __forceinline__ void wait_async() {
#if HAVE_ASYNC
#if __has_builtin(__builtin_amdgcn_s_wait_asynccnt)
  __builtin_amdgcn_s_wait_asynccnt(NWAIT);
#else
  asm volatile("s_wait_asynccnt %0" :: "i"(NWAIT) : "memory");
#endif
#endif
}

// split f32 -> bf16 hi + bf16 lo (native casts; compiler picks HW cvt)
__device__ __forceinline__ void bfsplit(float x, unsigned short& hi, unsigned short& lo) {
  __bf16 hb = (__bf16)x;
  float  hf = (float)hb;
  __bf16 lb = (__bf16)(x - hf);
  hi = __builtin_bit_cast(unsigned short, hb);
  lo = __builtin_bit_cast(unsigned short, lb);
}

// ============================================================
// K0: patient embedding  relu(patient@r1+b) @ r2 + b, l2norm
// ============================================================
__global__ void __launch_bounds__(256)
embed_kernel(const float* __restrict__ patient,
             const float* __restrict__ r1w, const float* __restrict__ r1b,
             const float* __restrict__ r2w, const float* __restrict__ r2b,
             float* __restrict__ pe) {
  __shared__ float prow[PD_];
  __shared__ float hid[H_];
  __shared__ float red[E_];
  const int b = blockIdx.x, t = threadIdx.x;
  if (t < PD_) prow[t] = patient[b * PD_ + t];
  __syncthreads();
  float s = r1b[t];
  for (int j = 0; j < PD_; ++j) s = fmaf(prow[j], r1w[j * H_ + t], s);
  hid[t] = fmaxf(s, 0.f);
  __syncthreads();
  float e = 0.f;
  if (t < E_) {
    e = r2b[t];
    for (int j = 0; j < H_; ++j) e = fmaf(hid[j], r2w[j * E_ + t], e);
    red[t] = e * e;
  }
  __syncthreads();
  for (int off = E_ / 2; off > 0; off >>= 1) {
    if (t < off) red[t] += red[t + off];
    __syncthreads();
  }
  if (t < E_) {
    float nrm = sqrtf(red[0]);
    pe[b * E_ + t] = e / fmaxf(nrm, 1e-12f);
  }
}

// ============================================================
// K1: async-pipelined similarity (bf16 split WMMA) + top-k
// ============================================================
__global__ void __launch_bounds__(256, 1)
sim_topk_kernel(const float* __restrict__ corpus,
                const float* __restrict__ pe,
                float* __restrict__ candScore,
                int* __restrict__ candIdx) {
  __shared__ float          bufA[DEPTH][TROWS][E_ + 4];   // f32 tile slots
  __shared__ unsigned short bhiS[TROWS][E_ + 8];          // bf16 hi
  __shared__ unsigned short bloS[TROWS][E_ + 8];          // bf16 lo
  __shared__ float          part[TROWS][8];
  __shared__ float          rnorm[TROWS];
  __shared__ float          trans[8][16][33];             // per-wave D transpose

  const int tid = threadIdx.x;
  const int w  = tid >> 5;        // wave -> patients [16w,16w+16)
  const int l  = tid & 31;
  const int ln = l & 15;
  const int lh = l >> 4;
  const int r8 = tid >> 3;        // staging/convert row (0..31)
  const int s8 = tid & 7;         // staging/convert seg (16 cols each)
  const long rowStart = (long)blockIdx.x * CHUNK;

  // ---- prime the async pipeline: tiles 0..DEPTH-2 ----
  for (int p = 0; p < DEPTH - 1 && p < TILES; ++p) {
    const float* src = corpus + (rowStart + p * TROWS + r8) * (long)E_ + s8 * 16;
    float* dst = &bufA[p][r8][s8 * 16];
#pragma unroll
    for (int i = 0; i < 4; ++i) gl2lds_b128(src + 4 * i, dst + 4 * i);
  }

  // ---- A fragments: 16 patients x E, hi/lo split, VGPR-resident ----
  // 16-bit A 16x32 layout: lane<16 holds M=lane, K={kh..kh+7, 16+kh..16+kh+7}
  v16u ahi[4], alo[4];
  {
    const int m  = w * 16 + ln;
    const int kh = lh ? 8 : 0;
    for (int kc = 0; kc < 4; ++kc) {
#pragma unroll
      for (int j = 0; j < 8; ++j) {
        unsigned short h, lo;
        bfsplit(pe[m * E_ + kc * 32 + kh + j], h, lo);
        ahi[kc][j] = h;  alo[kc][j] = lo;
        bfsplit(pe[m * E_ + kc * 32 + 16 + kh + j], h, lo);
        ahi[kc][8 + j] = h;  alo[kc][8 + j] = lo;
      }
    }
  }

  // running top-16: lane owns patient ln (wave w), column half lh
  float ts[16]; int ti[16];
#pragma unroll
  for (int q = 0; q < 16; ++q) { ts[q] = NEG_INF; ti[q] = -1; }
  float rmin = NEG_INF; int rslot = 0;

  for (int tIt = 0; tIt < TILES; ++tIt) {
    const long tileRow = rowStart + (long)tIt * TROWS;
    const int  slot    = tIt % DEPTH;

    // issue async loads for tile tIt+DEPTH-1
    if (tIt + DEPTH - 1 < TILES) {
      const int  pt  = tIt + DEPTH - 1;
      const float* src = corpus + (rowStart + (long)pt * TROWS + r8) * (long)E_ + s8 * 16;
      float* dst = &bufA[pt % DEPTH][r8][s8 * 16];
#pragma unroll
      for (int i = 0; i < 4; ++i) gl2lds_b128(src + 4 * i, dst + 4 * i);
    }
    wait_async<4 * (DEPTH - 1)>();   // oldest tile's 4 asyncs complete
    __syncthreads();                 // visibility + bhiS/bloS reuse fence

    // ---- cooperative f32 -> bf16 hi/lo conversion + row sumsq ----
    {
      const float4 x0 = *(const float4*)&bufA[slot][r8][s8 * 16 + 0];
      const float4 x1 = *(const float4*)&bufA[slot][r8][s8 * 16 + 4];
      const float4 x2 = *(const float4*)&bufA[slot][r8][s8 * 16 + 8];
      const float4 x3 = *(const float4*)&bufA[slot][r8][s8 * 16 + 12];
      float xs[16] = {x0.x,x0.y,x0.z,x0.w, x1.x,x1.y,x1.z,x1.w,
                      x2.x,x2.y,x2.z,x2.w, x3.x,x3.y,x3.z,x3.w};
      v8u hv0, hv1, lv0, lv1;
      float ssq = 0.f;
#pragma unroll
      for (int j = 0; j < 8; ++j) {
        unsigned short h, lo;
        bfsplit(xs[j], h, lo);
        hv0[j] = h; lv0[j] = lo;
        ssq = fmaf(xs[j], xs[j], ssq);
      }
#pragma unroll
      for (int j = 0; j < 8; ++j) {
        unsigned short h, lo;
        bfsplit(xs[8 + j], h, lo);
        hv1[j] = h; lv1[j] = lo;
        ssq = fmaf(xs[8 + j], xs[8 + j], ssq);
      }
      *(v8u*)&bhiS[r8][s8 * 16 + 0] = hv0;
      *(v8u*)&bhiS[r8][s8 * 16 + 8] = hv1;
      *(v8u*)&bloS[r8][s8 * 16 + 0] = lv0;
      *(v8u*)&bloS[r8][s8 * 16 + 8] = lv1;
      part[r8][s8] = ssq;
    }
    __syncthreads();
    if (tid < TROWS) {
      float s = 0.f;
#pragma unroll
      for (int g = 0; g < 8; ++g) s += part[tid][g];
      rnorm[tid] = 1.0f / fmaxf(sqrtf(s), 1e-12f);
    }
    __syncthreads();

    // ---- WMMA: D[16 patients x 32 corpus rows] as two 16x16 tiles ----
    v8f acc0 = {}, acc1 = {};
#pragma unroll
    for (int kc = 0; kc < 4; ++kc) {
      const int kb = kc * 32 + (lh ? 8 : 0);    // B: lane<16 holds K={kb..kb+7, kb+16..kb+23}
      v16bf Ah = __builtin_bit_cast(v16bf, ahi[kc]);
      v16bf Al = __builtin_bit_cast(v16bf, alo[kc]);
#pragma unroll
      for (int g = 0; g < 2; ++g) {
        const int n = g * 16 + ln;
        v8u h0 = *(const v8u*)&bhiS[n][kb];
        v8u h1 = *(const v8u*)&bhiS[n][kb + 16];
        v8u q0 = *(const v8u*)&bloS[n][kb];
        v8u q1 = *(const v8u*)&bloS[n][kb + 16];
        v16bf Bh = __builtin_bit_cast(v16bf, __builtin_shufflevector(
            h0, h1, 0,1,2,3,4,5,6,7,8,9,10,11,12,13,14,15));
        v16bf Bl = __builtin_bit_cast(v16bf, __builtin_shufflevector(
            q0, q1, 0,1,2,3,4,5,6,7,8,9,10,11,12,13,14,15));
        v8f& acc = g ? acc1 : acc0;
        // split precision: hi*hi + hi*lo + lo*hi  (~f32-accurate)
        acc = __builtin_amdgcn_wmma_f32_16x16x32_bf16(false, Ah, false, Bh, (short)0, acc, false, false);
        acc = __builtin_amdgcn_wmma_f32_16x16x32_bf16(false, Ah, false, Bl, (short)0, acc, false, false);
        acc = __builtin_amdgcn_wmma_f32_16x16x32_bf16(false, Al, false, Bh, (short)0, acc, false, false);
      }
    }

    // fold inverse norms (lane owns D columns ln and 16+ln), transpose
    {
      float rn0 = rnorm[ln], rn1 = rnorm[16 + ln];
#pragma unroll
      for (int r = 0; r < 8; ++r) {
        trans[w][r + lh * 8][ln]      = acc0[r] * rn0;
        trans[w][r + lh * 8][16 + ln] = acc1[r] * rn1;
      }
    }
    asm volatile("s_wait_dscnt 0" ::: "memory");   // same-wave DS in-order

    // per-lane top-k: patient ln, columns [16*lh, 16*lh+16)
#pragma unroll
    for (int j = 0; j < 16; ++j) {
      const int col = lh * 16 + j;
      const float sc = trans[w][ln][col];
      if (sc > rmin) {                              // rare after warmup
        const int gi = (int)(tileRow + col);
#pragma unroll
        for (int q = 0; q < 16; ++q) if (q == rslot) { ts[q] = sc; ti[q] = gi; }
        rmin = ts[0]; rslot = 0;
#pragma unroll
        for (int q = 1; q < 16; ++q) if (ts[q] < rmin) { rmin = ts[q]; rslot = q; }
      }
    }
  }

  // write 16 candidates per (block, patient, half)
  {
    const int p = w * 16 + ln;
    const long base = ((long)blockIdx.x * B_ + p) * CPP + lh * 16;
#pragma unroll
    for (int q = 0; q < 16; ++q) { candScore[base + q] = ts[q]; candIdx[base + q] = ti[q]; }
  }
}

// ============================================================
// K2: exact merge of per-block candidates -> sorted top-16
// ============================================================
__global__ void __launch_bounds__(256)
merge_topk_kernel(const float* __restrict__ candScore, const int* __restrict__ candIdx,
                  float* __restrict__ outScores, int* __restrict__ outIdx) {
  __shared__ float sc[POOL];
  __shared__ int   id[POOL];
  __shared__ float rsc[256];
  __shared__ int   rid[256];
  const int b = blockIdx.x, t = threadIdx.x;
  for (int i = 0; i < POOL / 256; ++i) {
    int j = t + i * 256;
    long src = ((long)(j / CPP) * B_ + b) * CPP + (j % CPP);
    sc[j] = candScore[src];
    id[j] = candIdx[src];
  }
  __syncthreads();
  for (int r = 0; r < K_; ++r) {
    float best = NEG_INF; int bslot = -1;
    for (int i = 0; i < POOL / 256; ++i) {
      int j = t + i * 256;
      float v = sc[j];
      if (v > best || (v == best && bslot >= 0 && id[j] >= 0 && id[j] < id[bslot])) {
        best = v; bslot = j;
      }
    }
    rsc[t] = best; rid[t] = bslot;
    __syncthreads();
    for (int off = 128; off > 0; off >>= 1) {
      if (t < off) {
        float a = rsc[t], c = rsc[t + off];
        int   ia = rid[t], ic = rid[t + off];
        bool take = (c > a) || (c == a && ic >= 0 && (ia < 0 || id[ic] < id[ia]));
        if (take) { rsc[t] = c; rid[t] = ic; }
      }
      __syncthreads();
    }
    if (t == 0) {
      int slot = rid[0];
      outScores[b * K_ + r] = rsc[0];
      outIdx[b * K_ + r]    = (slot >= 0) ? id[slot] : -1;
      if (slot >= 0) sc[slot] = NEG_INF;
    }
    __syncthreads();
  }
}

// ============================================================
// K3: gather + encoders + outcome + attribution (block/patient)
// ============================================================
__global__ void __launch_bounds__(256)
tail_kernel(const float* __restrict__ treatment, const float* __restrict__ confounders,
            const float* __restrict__ corpus,
            const float* __restrict__ tw,  const float* __restrict__ tb,
            const float* __restrict__ cw,  const float* __restrict__ cb2,
            const float* __restrict__ a1w, const float* __restrict__ a1b,
            const float* __restrict__ a2w, const float* __restrict__ a2b,
            const float* __restrict__ o1w, const float* __restrict__ o1b,
            const float* __restrict__ o2w, const float* __restrict__ o2b,
            const float* __restrict__ o3w, const float* __restrict__ o3b,
            const float* __restrict__ at1w, const float* __restrict__ at1b,
            const float* __restrict__ at2w, const float* __restrict__ at2b,
            float* __restrict__ dout) {
  __shared__ float flat[K_ * E_];
  __shared__ float comb[3 * H_];
  __shared__ float tmp[H_];
  __shared__ float o1[H_];
  __shared__ float o2[H_ / 2];
  __shared__ float ain[H_ + 1];
  __shared__ float red[128];
  __shared__ float trow[TD_];
  __shared__ float crow[CD_];
  __shared__ int   kidx[K_];
  __shared__ float att[K_];
  const int b = blockIdx.x, t = threadIdx.x;
  const int* gidx = (const int*)dout + OUT_IDX;
  if (t < K_)  kidx[t] = gidx[b * K_ + t];
  if (t < TD_) trow[t] = treatment[b * TD_ + t];
  if (t < CD_) crow[t] = confounders[b * CD_ + t];
  __syncthreads();
  for (int i = 0; i < K_ * E_; i += 256) {
    int j = i + t, k = j >> 7, c = j & 127;
    float v = corpus[(long)kidx[k] * E_ + c];
    flat[j] = v;
    dout[OUT_RETR + (long)b * K_ * E_ + j] = v;
  }
  __syncthreads();
  {
    float s = tb[t];
    for (int j = 0; j < TD_; ++j) s = fmaf(trow[j], tw[j * H_ + t], s);
    comb[t] = s;
    s = cb2[t];
    for (int j = 0; j < CD_; ++j) s = fmaf(crow[j], cw[j * H_ + t], s);
    comb[H_ + t] = s;
    s = a1b[t];
    for (int j = 0; j < K_ * E_; ++j) s = fmaf(flat[j], a1w[j * H_ + t], s);
    tmp[t] = fmaxf(s, 0.f);
  }
  __syncthreads();
  {
    float s = a2b[t];
    for (int j = 0; j < H_; ++j) s = fmaf(tmp[j], a2w[j * H_ + t], s);
    comb[2 * H_ + t] = s;
    ain[t] = s;
  }
  __syncthreads();
  {
    float s = o1b[t];
    for (int j = 0; j < 3 * H_; ++j) s = fmaf(comb[j], o1w[j * H_ + t], s);
    o1[t] = fmaxf(s, 0.f);
  }
  __syncthreads();
  if (t < H_ / 2) {
    float s = o2b[t];
    for (int j = 0; j < H_; ++j) s = fmaf(o1[j], o2w[j * (H_ / 2) + t], s);
    o2[t] = fmaxf(s, 0.f);
  }
  __syncthreads();
  if (t < 128) red[t] = o2[t] * o3w[t];
  __syncthreads();
  for (int off = 64; off > 0; off >>= 1) { if (t < off) red[t] += red[t + off]; __syncthreads(); }
  if (t == 0) {
    float out = red[0] + o3b[0];
    dout[OUT_OUTCOME + b] = out;
    ain[H_] = out;
  }
  __syncthreads();
  {
    float s = at1b[t];
    for (int j = 0; j < H_ + 1; ++j) s = fmaf(ain[j], at1w[j * H_ + t], s);
    tmp[t] = fmaxf(s, 0.f);
  }
  __syncthreads();
  if (t < K_) {
    float s = at2b[t];
    for (int j = 0; j < H_; ++j) s = fmaf(tmp[j], at2w[j * K_ + t], s);
    att[t] = s;
  }
  __syncthreads();
  if (t == 0) {
    float mx = att[0];
    for (int k = 1; k < K_; ++k) mx = fmaxf(mx, att[k]);
    float ssum = 0.f;
    for (int k = 0; k < K_; ++k) { float e = __expf(att[k] - mx); att[k] = e; ssum += e; }
    for (int k = 0; k < K_; ++k) dout[OUT_ATTR + b * K_ + k] = att[k] / ssum;
  }
}

// ============================================================
extern "C" void kernel_launch(void* const* d_in, const int* in_sizes, int n_in,
                              void* d_out, int out_size, void* d_ws, size_t ws_size,
                              hipStream_t stream) {
  const float* patient   = (const float*)d_in[0];
  const float* treatment = (const float*)d_in[1];
  const float* conf      = (const float*)d_in[2];
  const float* corpus    = (const float*)d_in[3];
  const float* r1w = (const float*)d_in[4];   const float* r1b = (const float*)d_in[5];
  const float* r2w = (const float*)d_in[6];   const float* r2b = (const float*)d_in[7];
  const float* tw  = (const float*)d_in[8];   const float* tb  = (const float*)d_in[9];
  const float* cw  = (const float*)d_in[10];  const float* cb2 = (const float*)d_in[11];
  const float* a1w = (const float*)d_in[12];  const float* a1b = (const float*)d_in[13];
  const float* a2w = (const float*)d_in[14];  const float* a2b = (const float*)d_in[15];
  const float* o1w = (const float*)d_in[16];  const float* o1b = (const float*)d_in[17];
  const float* o2w = (const float*)d_in[18];  const float* o2b = (const float*)d_in[19];
  const float* o3w = (const float*)d_in[20];  const float* o3b = (const float*)d_in[21];
  const float* at1w = (const float*)d_in[22]; const float* at1b = (const float*)d_in[23];
  const float* at2w = (const float*)d_in[24]; const float* at2b = (const float*)d_in[25];

  float* ws  = (float*)d_ws;
  float* pe  = ws + WS_PE;
  float* csc = ws + WS_CSC;
  int*   cid = (int*)(ws + WS_CID);
  float* out = (float*)d_out;

  embed_kernel<<<B_, 256, 0, stream>>>(patient, r1w, r1b, r2w, r2b, pe);
  sim_topk_kernel<<<NBLK, 256, 0, stream>>>(corpus, pe, csc, cid);
  merge_topk_kernel<<<B_, 256, 0, stream>>>(csc, cid, out + OUT_SCORES, (int*)out + OUT_IDX);
  tail_kernel<<<B_, 256, 0, stream>>>(treatment, conf, corpus,
                                      tw, tb, cw, cb2, a1w, a1b, a2w, a2b,
                                      o1w, o1b, o2w, o2b, o3w, o3b,
                                      at1w, at1b, at2w, at2b, out);
}
